// MLPAttention_54391465836710
// MI455X (gfx1250) — matrix operand order repaired and checked by reading, not compiled, not run on hardware
//
#include <hip/hip_runtime.h>
#include <hip/hip_bf16.h>

// MLP-attention for MI455X (gfx1250, wave32).
// Dominant work: dp = Wd(128x128) * dist(128x262144) GEMM done with
// v_wmma_f32_16x16x32_bf16 (bf16 inputs, f32 accumulate), fused with the
// relu/W2 reduction epilogue. dist (134 MB) is streamed exactly once.
// out = prob * value^T is a full-fp32 v_wmma_f32_16x16x4_f32 GEMM.

#define D_DIM   128
#define NQ      128
#define NK      2048
#define NCOL    (NQ * NK)      // 262144
#define W1S     384

typedef __attribute__((ext_vector_type(16))) __bf16 v16bf;
typedef __attribute__((ext_vector_type(8)))  float  v8f;
typedef __attribute__((ext_vector_type(2)))  float  v2f;

union F16Frag { v16bf v; uint4 q[2]; };

__device__ __forceinline__ unsigned short f32_to_bf16(float f) {
    unsigned int u = __float_as_uint(f);
    u += 0x7FFFu + ((u >> 16) & 1u);   // round-to-nearest-even
    return (unsigned short)(u >> 16);
}

// order-preserving float <-> uint keys for atomicMin
__device__ __forceinline__ unsigned int fkey(float f) {
    unsigned int u = __float_as_uint(f);
    return (u & 0x80000000u) ? ~u : (u | 0x80000000u);
}
__device__ __forceinline__ float fdecode(unsigned int k) {
    return (k & 0x80000000u) ? __uint_as_float(k ^ 0x80000000u)
                             : __uint_as_float(~k);
}

// ---------------------------------------------------------------------------
// Kernel 0: u[h][q] = b1[h] + Wq[h,:]·query[:,q] ; kp[h][k] = Wk[h,:]·key[:,k]
//           wd_bf16[h][d] = bf16(Wd[h][d]) ; init global-min key.
// ---------------------------------------------------------------------------
__global__ void __launch_bounds__(256) prep_kernel(
    const float* __restrict__ query, const float* __restrict__ key,
    const float* __restrict__ W1, const float* __restrict__ b1,
    float* __restrict__ ws_u, float* __restrict__ ws_kp,
    unsigned short* __restrict__ ws_wd, unsigned int* __restrict__ ws_min)
{
    int t = blockIdx.x * blockDim.x + threadIdx.x;
    if (t == 0) *ws_min = 0xFFFFFFFFu;

    if (t < D_DIM * NK) {                       // kp: 128 x 2048
        int h = t >> 11, k = t & (NK - 1);
        const float* wrow = W1 + h * W1S + D_DIM;   // Wk row
        float acc = 0.f;
        #pragma unroll 8
        for (int d = 0; d < D_DIM; ++d)
            acc = fmaf(wrow[d], key[d * NK + k], acc);
        ws_kp[h * NK + k] = acc;
    }
    if (t < D_DIM * NQ) {                       // u: 128 x 128
        int h = t >> 7, q = t & (NQ - 1);
        const float* wrow = W1 + h * W1S;           // Wq row
        float acc = b1[h];
        #pragma unroll 8
        for (int d = 0; d < D_DIM; ++d)
            acc = fmaf(wrow[d], query[d * NQ + q], acc);
        ws_u[h * NQ + q] = acc;
    }
    if (t < D_DIM * D_DIM) {                    // Wd -> bf16 row-major [h][d]
        int h = t >> 7, d = t & (D_DIM - 1);
        ws_wd[t] = f32_to_bf16(W1[h * W1S + 2 * D_DIM + d]);
    }
}

// ---------------------------------------------------------------------------
// Kernel 1: raw scores via WMMA. One workgroup = 8 waves handles a 128-column
// n-block (all 128 h rows). dist tile staged f32->bf16 into LDS transposed.
// ---------------------------------------------------------------------------
#define NB      128          // columns per block (one q row per block)
#define LDS_DS  136          // padded d-stride (bf16 elems) to dodge bank conflicts

__global__ void __launch_bounds__(256) scores_kernel(
    const float* __restrict__ dist,
    const float* __restrict__ ws_u, const float* __restrict__ ws_kp,
    const unsigned short* __restrict__ ws_wd,
    const float* __restrict__ W2, const float* __restrict__ b2,
    float* __restrict__ raw_scores, unsigned int* __restrict__ ws_min)
{
    __shared__ __align__(16) unsigned short ldsB[NB * LDS_DS]; // [n_local][d]
    __shared__ float sred[8 * NB];

    const int tid   = threadIdx.x;
    const int nBase = blockIdx.x * NB;

    // ---- stage dist[0:128][nBase:nBase+128] -> bf16, transposed into LDS ----
    #pragma unroll
    for (int i = 0; i < 16; ++i) {
        int c  = tid + 256 * i;          // 0..4095
        int d  = c >> 5;                 // 0..127
        int n4 = c & 31;                 // float4 index along n
        float4 v = *reinterpret_cast<const float4*>(dist + (size_t)d * NCOL + nBase + n4 * 4);
        int nl = n4 * 4;
        ldsB[(nl + 0) * LDS_DS + d] = f32_to_bf16(v.x);
        ldsB[(nl + 1) * LDS_DS + d] = f32_to_bf16(v.y);
        ldsB[(nl + 2) * LDS_DS + d] = f32_to_bf16(v.z);
        ldsB[(nl + 3) * LDS_DS + d] = f32_to_bf16(v.w);
    }
    __syncthreads();

    const int wave  = tid >> 5;
    const int lane  = tid & 31;
    const int half  = (lane < 16) ? 0 : 1;
    const int hBase = wave * 16;
    const int moff  = half ? 8 : 0;
    const int qBlk  = nBase >> 11;       // whole block is inside one q row
    const int kBase = nBase & (NK - 1);

    // ---- A fragments (Wd rows for this wave's 16 h), resident all kernel ----
    F16Frag aF[4];
    {
        int row = hBase + (lane & 15);
        const unsigned short* wr = ws_wd + row * D_DIM;
        #pragma unroll
        for (int s = 0; s < 4; ++s) {
            int d0 = 32 * s + (half ? 8 : 0);
            aF[s].q[0] = *reinterpret_cast<const uint4*>(wr + d0);
            aF[s].q[1] = *reinterpret_cast<const uint4*>(wr + d0 + 16);
        }
    }
    // loop-invariant per-h terms: W2[h] and u[h][qBlk] (+b1 already folded)
    float w2v[8], uv[8];
    #pragma unroll
    for (int j = 0; j < 8; ++j) {
        int h  = hBase + moff + j;
        w2v[j] = W2[h];
        uv[j]  = ws_u[h * NQ + qBlk];
    }

    float pk[8];

    #pragma unroll 2
    for (int ct = 0; ct < 8; ++ct) {
        v8f c = {};
        const int nl = ct * 16 + (lane & 15);
        #pragma unroll
        for (int s = 0; s < 4; ++s) {
            F16Frag bF;
            const unsigned short* p = &ldsB[nl * LDS_DS + 32 * s + (half ? 8 : 0)];
            bF.q[0] = *reinterpret_cast<const uint4*>(p);
            bF.q[1] = *reinterpret_cast<const uint4*>(p + 16);
            c = __builtin_amdgcn_wmma_f32_16x16x32_bf16(
                    false, aF[s].v, false, bF.v, (short)0, c, false, false);
        }
        // fused epilogue: relu(dp + u + kp) weighted by W2, reduce 8 h locally
        const int k = kBase + nl;
        float pp = 0.f;
        #pragma unroll
        for (int j = 0; j < 8; ++j) {
            int h = hBase + moff + j;
            float x = c[j] + uv[j] + ws_kp[h * NK + k];
            pp = fmaf(w2v[j], fmaxf(x, 0.f), pp);
        }
        pp += __shfl_xor(pp, 16, 32);     // combine the two 8-h halves
        pk[ct] = pp;
    }

    if (lane < 16) {
        #pragma unroll
        for (int ct = 0; ct < 8; ++ct)
            sred[wave * NB + ct * 16 + lane] = pk[ct];
    }
    __syncthreads();

    // cross-wave h reduction + raw score + block min
    float s = 0.f;
    if (tid < NB) {
        s = b2[0];
        #pragma unroll
        for (int w = 0; w < 8; ++w) s += sred[w * NB + tid];
        raw_scores[nBase + tid] = s;
    }
    __syncthreads();
    float* red = sred;                    // reuse as 128-entry min buffer
    if (tid < NB) red[tid] = s;
    __syncthreads();
    for (int off = 64; off >= 1; off >>= 1) {
        if (tid < off) red[tid] = fminf(red[tid], red[tid + off]);
        __syncthreads();
    }
    if (tid == 0) atomicMin(ws_min, fkey(red[0]));
}

// ---------------------------------------------------------------------------
// Kernel 2: mask + write final scores + row softmax -> normalized probs to ws.
// One workgroup per q row.
// ---------------------------------------------------------------------------
__global__ void __launch_bounds__(256) softmax_kernel(
    float* __restrict__ scores,                 // in: raw, out: masked (in place)
    const unsigned char* __restrict__ mask,     // jax bool = 1 byte
    const unsigned int* __restrict__ ws_min,
    float* __restrict__ ws_prob)
{
    __shared__ float sprob[NK];
    __shared__ float red[256];

    const int tid = threadIdx.x;
    const int q   = blockIdx.x;
    const float neg = fdecode(*ws_min) - 20.0f;

    float lmax = -3.0e38f;
    for (int k = tid; k < NK; k += 256) {
        float s = scores[q * NK + k];
        if (!mask[q * NK + k]) s += neg;
        scores[q * NK + k] = s;           // returned (masked) scores
        sprob[k] = s;
        lmax = fmaxf(lmax, s);
    }
    red[tid] = lmax; __syncthreads();
    for (int off = 128; off >= 1; off >>= 1) {
        if (tid < off) red[tid] = fmaxf(red[tid], red[tid + off]);
        __syncthreads();
    }
    const float rowmax = red[0];
    __syncthreads();

    float lsum = 0.f;
    for (int k = tid; k < NK; k += 256) {
        float e = __expf(sprob[k] - rowmax);
        sprob[k] = e;
        lsum += e;
    }
    red[tid] = lsum; __syncthreads();
    for (int off = 128; off >= 1; off >>= 1) {
        if (tid < off) red[tid] += red[tid + off];
        __syncthreads();
    }
    const float inv = 1.0f / red[0];
    __syncthreads();

    for (int k = tid; k < NK; k += 256)
        ws_prob[q * NK + k] = sprob[k] * inv;
}

// ---------------------------------------------------------------------------
// Kernel 3: out[d][q] = sum_k prob[q][k] * value[d][k], full-fp32 WMMA GEMM.
// C = prob(128x2048) x value^T(2048x128): 64 16x16 tiles, 1 tile per wave.
// Two interleaved accumulators break the WMMA->WMMA RAW chain.
// ---------------------------------------------------------------------------
__global__ void __launch_bounds__(256) out_gemm_kernel(
    const float* __restrict__ prob, const float* __restrict__ value,
    float* __restrict__ out)
{
    const int tid  = threadIdx.x;
    const int wave = tid >> 5;
    const int lane = tid & 31;
    const int id    = blockIdx.x * 8 + wave;   // 0..63
    const int mBase = (id >> 3) * 16;          // q-tile
    const int nBase = (id & 7) * 16;           // d-tile
    const int koff  = (lane < 16) ? 0 : 2;
    const int moff  = (lane < 16) ? 0 : 8;

    const float* pRow = prob  + (size_t)(mBase + (lane & 15)) * NK + koff;
    const float* vRow = value + (size_t)(nBase + (lane & 15)) * NK + koff;

#if __has_builtin(__builtin_amdgcn_wmma_f32_16x16x4_f32)
    v8f c0 = {}, c1 = {};
    #pragma unroll 4
    for (int k = 0; k < NK; k += 8) {
        // A 16x4 f32: lanes 0-15 hold K0,K1; lanes 16-31 hold K2,K3 (per ISA)
        v2f a0 = *reinterpret_cast<const v2f*>(pRow + k);
        v2f b0 = *reinterpret_cast<const v2f*>(vRow + k);
        c0 = __builtin_amdgcn_wmma_f32_16x16x4_f32(
                 false, a0, false, b0, (short)0, c0, false, false);
        v2f a1 = *reinterpret_cast<const v2f*>(pRow + k + 4);
        v2f b1 = *reinterpret_cast<const v2f*>(vRow + k + 4);
        c1 = __builtin_amdgcn_wmma_f32_16x16x4_f32(
                 false, a1, false, b1, (short)0, c1, false, false);
    }
    #pragma unroll
    for (int j = 0; j < 8; ++j) {
        int q = mBase + moff + j;
        int d = nBase + (lane & 15);
        out[d * NQ + q] = c0[j] + c1[j];
    }
#else
    // scalar fallback (should not be hit on gfx1250)
    float acc[8] = {};
    for (int k = 0; k < NK; ++k) {
        float v = value[(nBase + (lane & 15)) * NK + k];
        #pragma unroll
        for (int j = 0; j < 8; ++j)
            acc[j] = fmaf(prob[(mBase + moff + j) * NK + k], v, acc[j]);
    }
    #pragma unroll
    for (int j = 0; j < 8; ++j)
        out[(nBase + (lane & 15)) * NQ + mBase + moff + j] = acc[j];
#endif
}

// ---------------------------------------------------------------------------
extern "C" void kernel_launch(void* const* d_in, const int* in_sizes, int n_in,
                              void* d_out, int out_size, void* d_ws, size_t ws_size,
                              hipStream_t stream) {
    const float* query = (const float*)d_in[0];
    const float* key   = (const float*)d_in[1];
    const float* value = (const float*)d_in[2];
    const float* dist  = (const float*)d_in[3];
    const unsigned char* mask = (const unsigned char*)d_in[4];
    const float* W1    = (const float*)d_in[5];
    const float* b1    = (const float*)d_in[6];
    const float* W2    = (const float*)d_in[7];
    const float* b2    = (const float*)d_in[8];

    float* out    = (float*)d_out;            // 128*128
    float* scores = out + D_DIM * NQ;         // 128*2048 (masked scores)

    // workspace layout (~2.2 MB)
    char* ws = (char*)d_ws;
    float*          ws_u    = (float*)ws;                              // 64 KB
    float*          ws_kp   = (float*)(ws + 65536);                    // 1 MB
    unsigned short* ws_wd   = (unsigned short*)(ws + 1114112);         // 32 KB
    unsigned int*   ws_min  = (unsigned int*)(ws + 1146880);           // 4 B
    float*          ws_prob = (float*)(ws + 1147136);                  // 1 MB

    prep_kernel<<<1024, 256, 0, stream>>>(query, key, W1, b1,
                                          ws_u, ws_kp, ws_wd, ws_min);
    scores_kernel<<<NCOL / NB, 256, 0, stream>>>(dist, ws_u, ws_kp, ws_wd,
                                                 W2, b2, scores, ws_min);
    softmax_kernel<<<NQ, 256, 0, stream>>>(scores, mask, ws_min, ws_prob);
    out_gemm_kernel<<<8, 256, 0, stream>>>(ws_prob, value, out);
}
